// PushAFMModel_8289286881327
// MI455X (gfx1250) — matrix operand order repaired and checked by reading, not compiled, not run on hardware
//
#include <hip/hip_runtime.h>
#include <hip/hip_bf16.h>
#include <math.h>

// ---------------------------------------------------------------------------
// Fused AFM forward for MI455X (gfx1250, wave32).
// One block (256 thr = 8 waves) per batch row. All intermediates in LDS:
//   x (20x64 f32), x_cross (192x64 f16 padded), W1 (64x64 f16).
// Pair-interaction MLP h = relu(xc @ W1^T + b1) on v_wmma_f32_16x16x32_f16;
// bias/relu/dot(w2) fused on the accumulator fragments (h never materialized).
// Roofline: embedding gathers (~450 MB total) dominate -> ~20us @ 23.3 TB/s;
// the 12.75 GFLOP interaction matmul in f16 WMMA is a few us. So this round
// widens all gathers to b128 and removes branch-gated loads to keep the
// memory pipe full; WMMA section unchanged (codegen already optimal).
// ---------------------------------------------------------------------------

typedef _Float16 v16h __attribute__((ext_vector_type(16)));
typedef _Float16 v8h  __attribute__((ext_vector_type(8)));
typedef float    v8f  __attribute__((ext_vector_type(8)));

static constexpr int kNS = 16;      // sparse fields
static constexpr int kNV = 4;       // varlen fields
static constexpr int kF  = 20;      // total fields
static constexpr int kV  = 100000;  // vocab (tables have kV+1 rows; row 0 = zeros)
static constexpr int kD  = 64;      // emb dim
static constexpr int kL  = 50;      // varlen history
static constexpr int kA  = 64;      // attn dim
static constexpr int kP  = 190;     // C(20,2) pairs
static constexpr int kPP = 192;     // pairs padded to WMMA M multiple

__global__ __launch_bounds__(256)
void afm_fused_kernel(const int*   __restrict__ sparse_ids,   // (B,16)
                      const int*   __restrict__ varlen_ids,   // (B,4,50)
                      const float* __restrict__ sparse_emb,   // (16,V+1,64)
                      const float* __restrict__ varlen_emb,   // (4,V+1,64)
                      const float* __restrict__ attn_w1,      // (64,64)
                      const float* __restrict__ attn_b1,      // (64,)
                      const float* __restrict__ attn_w2,      // (64,)
                      const float* __restrict__ fc_w,         // (1,64)
                      const float* __restrict__ fc_b,         // (1,)
                      float*       __restrict__ out)          // (B,)
{
    __shared__ __align__(16) float xf[kF][kD];         // 5 KB
    __shared__ __align__(16) _Float16 xcs[kPP][kD];    // 24 KB
    __shared__ __align__(16) _Float16 w1h[kA][kD];     // 8 KB
    __shared__ float b1s[kA], w2s[kA];
    __shared__ float scoreS[kPP];
    __shared__ float red[256];
    __shared__ float fpart[4][kD];
    __shared__ unsigned char cis[kP], cjs[kP];
    __shared__ int  sids[kNS];
    __shared__ int  vids[kNV * kL];
    __shared__ __align__(16) float4 vpool[kNV][4][kD / 4];  // 4 KB
    __shared__ int  cpool[kNV][4];

    const int b   = blockIdx.x;
    const int tid = threadIdx.x;

    // ---- phase 0: ids -> LDS, pair table, W1 -> f16, bias/w2 ---------------
    if (tid < kNV * kL) vids[tid] = varlen_ids[(long)b * kNV * kL + tid];
    if (tid >= 224 && tid < 224 + kNS) sids[tid - 224] = sparse_ids[b * kNS + (tid - 224)];
    if (tid < kP) {
        int rem = tid, i = 0;
        while (rem >= (kF - 1 - i)) { rem -= (kF - 1 - i); ++i; }
        cis[tid] = (unsigned char)i;
        cjs[tid] = (unsigned char)(i + 1 + rem);
    }
    for (int idx = tid; idx < kA * kD; idx += 256)
        w1h[idx >> 6][idx & 63] = (_Float16)attn_w1[idx];
    if (tid < kA) { b1s[tid] = attn_b1[tid]; w2s[tid] = attn_w2[tid]; }
    __syncthreads();

    // ---- phase 1a: sparse gather, one b128 per thread ----------------------
    {   // 16 fields x 16 float4 chunks == 256 threads
        const int fld = tid >> 4, c = tid & 15;
        const float4* srow = (const float4*)(sparse_emb +
                             ((long)fld * (kV + 1) + (long)sids[fld]) * kD);
        ((float4*)&xf[fld][0])[c] = srow[c];
    }
    // ---- phase 1b: varlen gather (b128, unconditional loads: row 0 = zeros)
    {   // fld(4) x sub(4 l-subsets) x chunk(16): lanes 0-15 of a wave read one
        // full 256B row contiguously per step -> coalesced global_load_b128.
        const int fld = tid >> 6;
        const int sub = (tid >> 4) & 3;
        const int c   = tid & 15;
        const float4* tab4 = (const float4*)(varlen_emb +
                             (long)fld * (kV + 1) * kD) + c;
        float4 acc = make_float4(0.f, 0.f, 0.f, 0.f);
        int cnt = 0;
        for (int l = sub; l < kL; l += 4) {
            const int id = vids[fld * kL + l];
            const float4 v = tab4[(long)id * (kD / 4)];
            acc.x += v.x; acc.y += v.y; acc.z += v.z; acc.w += v.w;
            cnt += (id != 0);
        }
        vpool[fld][sub][c] = acc;
        if (c == 0) cpool[fld][sub] = cnt;
    }
    __syncthreads();
    if (tid < kNV * (kD / 4)) {   // 64 threads finalize the mean-pool
        const int fld = tid >> 4, c = tid & 15;
        const float4 a0 = vpool[fld][0][c], a1 = vpool[fld][1][c];
        const float4 a2 = vpool[fld][2][c], a3 = vpool[fld][3][c];
        const float cnt = (float)(cpool[fld][0] + cpool[fld][1] +
                                  cpool[fld][2] + cpool[fld][3]);
        const float inv = 1.f / (cnt + 1e-8f);
        float4 r;
        r.x = (a0.x + a1.x + a2.x + a3.x) * inv;
        r.y = (a0.y + a1.y + a2.y + a3.y) * inv;
        r.z = (a0.z + a1.z + a2.z + a3.z) * inv;
        r.w = (a0.w + a1.w + a2.w + a3.w) * inv;
        ((float4*)&xf[kNS + fld][0])[c] = r;
    }
    __syncthreads();

    // ---- phase 2: pairwise cross products -> f16 in LDS --------------------
    for (int idx = tid; idx < kP * kD; idx += 256) {
        const int p = idx >> 6, d = idx & 63;
        xcs[p][d] = (_Float16)(xf[cis[p]][d] * xf[cjs[p]][d]);
    }
    if (tid < 2 * kD) xcs[kP + (tid >> 6)][tid & 63] = (_Float16)0.f;  // pad rows
    __syncthreads();

    // ---- phase 3: WMMA MLP + fused bias/relu/dot(w2) -> scoreS -------------
    {
        const int  lane = tid & 31;
        const int  wave = tid >> 5;
        const bool lo   = lane < 16;        // half-wave select per ISA layouts
        const int  nrow = lane & 15;        // M (A/C frag) and N (B/C frag) in tile
        const int  nm   = (wave < 4) ? 2 : 1;   // waves 0-3 do 2 M-tiles
        for (int t = 0; t < nm; ++t) {
            const int mtile = wave + t * 8;     // covers M-tiles 0..11
            float mpart[8];
            #pragma unroll
            for (int r = 0; r < 8; ++r) mpart[r] = 0.f;
            #pragma unroll
            for (int ntile = 0; ntile < 4; ++ntile) {
                v8f c = {};
                #pragma unroll
                for (int kt = 0; kt < 2; ++kt) {       // K = 64 -> 2 chained WMMAs
                    const int ktile = kt * 32;
                    // A (16x32 f16): lanes<16 hold K[kt..+7],K[kt+16..+23];
                    //                lanes>=16 hold K[kt+8..+15],K[kt+24..+31]
                    const _Float16* arow = &xcs[mtile * 16 + nrow][0];
                    const int ka = ktile + (lo ? 0 : 8);
                    v8h a0 = *(const v8h*)(arow + ka);
                    v8h a1 = *(const v8h*)(arow + ka + 16);
                    v16h av = __builtin_shufflevector(a0, a1,
                              0,1,2,3,4,5,6,7,8,9,10,11,12,13,14,15);
                    // B (32x16 f16): N = lane&15; lanes<16 K[kt..+7],K[kt+16..+23];
                    // lanes>=16 K[kt+8..+15],K[kt+24..+31]. B[k][n] = W1[n][k] (h = xc @ W1^T)
                    const _Float16* brow = &w1h[ntile * 16 + nrow][0];
                    const int kb = ktile + (lo ? 0 : 8);
                    v8h b0  = *(const v8h*)(brow + kb);
                    v8h b1v = *(const v8h*)(brow + kb + 16);
                    v16h bv = __builtin_shufflevector(b0, b1v,
                              0,1,2,3,4,5,6,7,8,9,10,11,12,13,14,15);
                    c = __builtin_amdgcn_wmma_f32_16x16x32_f16(
                            false, av, false, bv, (short)0, c, false, false);
                }
                // C/D layout: VGPR r -> M = r + (lo?0:8), N = lane&15
                const int   aidx = ntile * 16 + nrow;
                const float bb = b1s[aidx], ww = w2s[aidx];
                #pragma unroll
                for (int r = 0; r < 8; ++r) {
                    float h = fmaxf(c[r] + bb, 0.f);    // bias + relu
                    mpart[r] = fmaf(h, ww, mpart[r]);   // dot with w2
                }
            }
            // reduce over the 16 lanes sharing an M row (each half-wave group)
            #pragma unroll
            for (int r = 0; r < 8; ++r) {
                float v = mpart[r];
                v += __shfl_xor(v, 8, 16);
                v += __shfl_xor(v, 4, 16);
                v += __shfl_xor(v, 2, 16);
                v += __shfl_xor(v, 1, 16);
                if (nrow == 0) scoreS[mtile * 16 + r + (lo ? 0 : 8)] = v;
            }
        }
    }
    __syncthreads();

    // ---- phase 4: softmax over 190 pairs -----------------------------------
    const float sc = (tid < kP) ? scoreS[tid] : -3.0e38f;
    red[tid] = sc; __syncthreads();
    for (int s = 128; s > 0; s >>= 1) {
        if (tid < s) red[tid] = fmaxf(red[tid], red[tid + s]);
        __syncthreads();
    }
    const float gmax = red[0]; __syncthreads();
    float e = 0.f;
    if (tid < kP) { e = __expf(sc - gmax); scoreS[tid] = e; }
    red[tid] = e; __syncthreads();
    for (int s = 128; s > 0; s >>= 1) {
        if (tid < s) red[tid] += red[tid + s];
        __syncthreads();
    }
    const float invsum = 1.f / red[0];
    __syncthreads();

    // ---- phase 5: f[d] = sum_p attn[p]*xc[p][d]; logits; sigmoid -----------
    {
        const int d = tid & 63, g = tid >> 6;
        float acc = 0.f;
        for (int p = g; p < kP; p += 4)
            acc += scoreS[p] * (float)xcs[p][d];
        fpart[g][d] = acc;
    }
    __syncthreads();
    if (tid < kD) {
        const float fd = (fpart[0][tid] + fpart[1][tid] +
                          fpart[2][tid] + fpart[3][tid]) * invsum;
        red[tid] = fd * fc_w[tid];
    }
    __syncthreads();
    if (tid < 32) red[tid] += red[tid + 32];
    __syncthreads();
    if (tid == 0) {
        float z = 0.f;
        #pragma unroll
        for (int i = 0; i < 32; ++i) z += red[i];
        z += fc_b[0];
        out[b] = 1.f / (1.f + __expf(-z));
    }
}

extern "C" void kernel_launch(void* const* d_in, const int* in_sizes, int n_in,
                              void* d_out, int out_size, void* d_ws, size_t ws_size,
                              hipStream_t stream) {
    const int*   sparse_ids = (const int*)  d_in[0];
    const int*   varlen_ids = (const int*)  d_in[1];
    const float* sparse_emb = (const float*)d_in[2];
    const float* varlen_emb = (const float*)d_in[3];
    const float* attn_w1    = (const float*)d_in[4];
    const float* attn_b1    = (const float*)d_in[5];
    const float* attn_w2    = (const float*)d_in[6];
    const float* fc_w       = (const float*)d_in[7];
    const float* fc_b       = (const float*)d_in[8];
    float*       out        = (float*)d_out;

    const int B = in_sizes[0] / kNS;   // 8192
    afm_fused_kernel<<<dim3(B), dim3(256), 0, stream>>>(
        sparse_ids, varlen_ids, sparse_emb, varlen_emb,
        attn_w1, attn_b1, attn_w2, fc_w, fc_b, out);
}
